// Encoder_24215025615261
// MI455X (gfx1250) — matrix operand order repaired and checked by reading, not compiled, not run on hardware
//
#include <hip/hip_runtime.h>

typedef _Float16 half_t;
typedef __attribute__((ext_vector_type(16))) _Float16 v16h;
typedef __attribute__((ext_vector_type(8)))  float    v8f;

#define WMMA_F16(a, b, c) \
  __builtin_amdgcn_wmma_f32_16x16x32_f16(false, (a), false, (b), (short)0, (c), false, false)

// Flip to 0 if the async-to-LDS asm ever fails to assemble.
#define USE_ASYNC_LDS 1

static __device__ __forceinline__ v8f zero8() {
  v8f z = {0.f, 0.f, 0.f, 0.f, 0.f, 0.f, 0.f, 0.f};
  return z;
}

// Low 32 bits of a flat pointer to __shared__ = LDS byte offset (aperture layout).
static __device__ __forceinline__ unsigned lds_addr32(const void* p) {
  return (unsigned)(size_t)p;
}

// Async DMA: 16 bytes global -> LDS per lane (ASYNCcnt-tracked).
static __device__ __forceinline__ void async_g2l_b128(const void* gsrc, void* ldst) {
#if USE_ASYNC_LDS
  asm volatile("global_load_async_to_lds_b128 %0, %1, off"
               :: "v"(lds_addr32(ldst)), "v"(gsrc)
               : "memory");
#else
  *(int4*)ldst = *(const int4*)gsrc;
#endif
}

static __device__ __forceinline__ void wait_asynccnt0() {
#if USE_ASYNC_LDS
  asm volatile("s_wait_asynccnt 0" ::: "memory");
#endif
}

// ---- Fragment index helpers (CDNA5 ISA 7.12.2, wave32) ----
// A (16x32 f16, MxK): lanes 0-15: M=lane, K 0-7 (+16 in VGPRs 4-7); lanes 16-31: K+8.
static __device__ __forceinline__ int afrag_k(int g, int p, int lane) {
  return ((g & 4) ? 16 : 0) + ((lane & 16) ? 8 : 0) + 2 * (g & 3) + p;
}
// B (32x16 f16, KxN): lane&15 = N; lanes 0-15: K 0-7 (+16 in VGPRs 4-7); lanes 16-31: K+8.
static __device__ __forceinline__ int bfrag_k(int g, int p, int lane) {
  return ((g & 4) ? 16 : 0) + ((lane & 16) ? 8 : 0) + 2 * (g & 3) + p;
}

// Load A fragment: element (m,k) at p[m*sm + k*sk]
static __device__ __forceinline__ v16h load_a_strided_h(const half_t* p, long sm, long sk) {
  const int lane = threadIdx.x & 31;
  const int m = lane & 15;
  v16h a;
#pragma unroll
  for (int g = 0; g < 8; ++g) {
#pragma unroll
    for (int q = 0; q < 2; ++q) {
      const int k = afrag_k(g, q, lane);
      a[2 * g + q] = p[(long)m * sm + (long)k * sk];
    }
  }
  return a;
}

// Load B fragment: element (k,n) at p[k*sk + n*sn]
static __device__ __forceinline__ v16h load_b_strided_h(const half_t* p, long sk, long sn) {
  const int lane = threadIdx.x & 31;
  const int n = lane & 15;
  v16h b;
#pragma unroll
  for (int g = 0; g < 8; ++g) {
#pragma unroll
    for (int q = 0; q < 2; ++q) {
      const int k = bfrag_k(g, q, lane);
      b[2 * g + q] = p[(long)k * sk + (long)n * sn];
    }
  }
  return b;
}

// B fragment from the staged conv tile Bs[32][136]; col c<0 redirects to the
// left-edge column stored at index 129.
static __device__ __forceinline__ v16h load_b_tile(const half_t* bs, int cbase) {
  const int lane = threadIdx.x & 31;
  int c = cbase + (lane & 15);
  if (c < 0) c = 129;
  v16h b;
#pragma unroll
  for (int g = 0; g < 8; ++g) {
#pragma unroll
    for (int q = 0; q < 2; ++q) {
      const int k = bfrag_k(g, q, lane);
      b[2 * g + q] = bs[(long)k * 136 + c];
    }
  }
  return b;
}

// B fragment from f32 row-major [rows, 64] with row bound (rel embeddings)
static __device__ __forceinline__ v16h load_b_f32rows(const float* er, int k0, int jbase, int jmax) {
  const int lane = threadIdx.x & 31;
  const int j = jbase + (lane & 15);
  v16h b;
#pragma unroll
  for (int g = 0; g < 8; ++g) {
#pragma unroll
    for (int q = 0; q < 2; ++q) {
      const int k = k0 + bfrag_k(g, q, lane);
      b[2 * g + q] = (j < jmax) ? (half_t)er[(long)j * 64 + k] : (half_t)0.f;
    }
  }
  return b;
}

// ======================================================================
// Tiled WMMA GEMM / conv1d(k=3):
//   Y[b,m,n] = act( (sum_{tap,k} W[m,k,tap] * X[b,k,n+tap-1] + bias[m]) * scale )
// W f32 (converted to f16 into LDS), X f16 [B,K,N].
// Block tile 64(M) x 128(N), BK=32; 8 waves as 2(M) x 4(N); wave tile 32x32.
// B tile staged ONCE per k-step for all 3 taps (cols n0-1 .. n0+128).
// Interior B copy uses gfx1250 async global->LDS DMA.
// ======================================================================
template <int TAPS, bool RELU, bool OUT16, bool MASKOUT>
__global__ __launch_bounds__(256) void gemm_wmma(
    const float* __restrict__ W, const half_t* __restrict__ X,
    const float* __restrict__ bias, const float* __restrict__ mask,
    half_t* __restrict__ Y16, float* __restrict__ Y32,
    int M, int K, int N, float scale) {
  __shared__ half_t As[TAPS][64][40];   // [tap][BM][BK+pad]
  __shared__ half_t Bs[32][136];        // cols 0..127 -> X[n0..n0+127]; 128 -> X[n0+128]; 129 -> X[n0-1]
  const int tid = threadIdx.x;
  const int b = blockIdx.z;
  const int m0 = blockIdx.y * 64;
  const int n0 = blockIdx.x * 128;
  const int w = tid >> 5;
  const int lane = tid & 31;
  const int wm = w >> 2;                // 0..1
  const int wn = w & 3;                 // 0..3
  const int grp = lane >> 4, nc = lane & 15;

  v8f acc[2][2] = {{zero8(), zero8()}, {zero8(), zero8()}};

  for (int k0 = 0; k0 < K; k0 += 32) {
    __syncthreads();
    // ---- stage A: TAPS x 64 x 32, f32 -> f16 ----
#pragma unroll 2
    for (int idx = tid; idx < TAPS * 64 * 32; idx += 256) {
      const int tap = idx >> 11;
      const int r = idx & 2047;
      const int i = r >> 5, kk = r & 31;
      As[tap][i][kk] = (half_t)W[((size_t)(m0 + i) * K + (k0 + kk)) * TAPS + tap];
    }
    // prefetch next W panel rows
    if (k0 + 32 < K) {
      const int i = tid & 63;
      __builtin_prefetch(&W[((size_t)(m0 + i) * K + (k0 + 32)) * TAPS], 0, 1);
    }
    // ---- edge columns for conv taps (DS stores, disjoint from async region) ----
    if (TAPS == 3 && tid < 32) {
      const size_t row = (size_t)b * K + (k0 + tid);
      const int nl = n0 - 1, nr = n0 + 128;
      Bs[tid][129] = (nl >= 0) ? X[row * N + nl] : (half_t)0.f;
      Bs[tid][128] = (nr < N) ? X[row * N + nr] : (half_t)0.f;
    }
    // ---- stage B interior (32 x 128 halves) via async DMA: 2 x 16B per thread ----
#pragma unroll
    for (int t2 = 0; t2 < 512; t2 += 256) {
      const int idx = tid + t2;
      const int r = idx >> 4, ch = idx & 15;
      const half_t* src = X + ((size_t)b * K + (k0 + r)) * N + n0 + ch * 8;
      async_g2l_b128(src, &Bs[r][ch * 8]);
    }
    wait_asynccnt0();
    __syncthreads();

    // ---- compute: per tap, 2 A-frags x 2 B-frags -> 4 WMMAs ----
#pragma unroll
    for (int tap = 0; tap < TAPS; ++tap) {
      v16h a0 = load_a_strided_h(&As[tap][wm * 32][0], 40, 1);
      v16h a1 = load_a_strided_h(&As[tap][wm * 32 + 16][0], 40, 1);
      const int cbase = wn * 32 + ((TAPS == 3) ? (tap - 1) : 0);
      v16h b0 = load_b_tile(&Bs[0][0], cbase);
      v16h b1 = load_b_tile(&Bs[0][0], cbase + 16);
      acc[0][0] = WMMA_F16(a0, b0, acc[0][0]);
      acc[0][1] = WMMA_F16(a0, b1, acc[0][1]);
      acc[1][0] = WMMA_F16(a1, b0, acc[1][0]);
      acc[1][1] = WMMA_F16(a1, b1, acc[1][1]);
    }
  }

  // ---- epilogue (C layout: VGPR g -> M=g / g+8 by lane group, N = lane&15) ----
#pragma unroll
  for (int at = 0; at < 2; ++at) {
#pragma unroll
    for (int g = 0; g < 8; ++g) {
      const int m = m0 + wm * 32 + at * 16 + g + grp * 8;
      const float bv = bias[m];
#pragma unroll
      for (int bt = 0; bt < 2; ++bt) {
        const int n = n0 + wn * 32 + bt * 16 + nc;
        float v = (acc[at][bt][g] + bv) * scale;
        if (RELU) v = fmaxf(v, 0.f);
        if (MASKOUT) v *= mask[(size_t)b * N + n];
        const size_t off = ((size_t)b * M + m) * N + n;
        if (OUT16) Y16[off] = (half_t)v;
        else       Y32[off] = v;
      }
    }
  }
}

// ======================================================================
// Windowed relative-position attention (window ±10).
// One wave per (b, h, 16-row tile); q/k/v f16 in [B, 512, T] layout.
// 1/sqrt(Dk) folded into Q projection.
// ======================================================================
__global__ __launch_bounds__(32) void attn_wmma(
    const half_t* __restrict__ qb, const half_t* __restrict__ kb,
    const half_t* __restrict__ vb, const float* __restrict__ erk,
    const float* __restrict__ erv, const float* __restrict__ mask,
    half_t* __restrict__ ob, int T) {
  __shared__ float  S[16][512];
  __shared__ half_t P[16][512];
  __shared__ float  RB[16][32];
  __shared__ float  EV[21][64];
  const int lane = threadIdx.x;
  const int bh = blockIdx.y;
  const int b = bh >> 3, h = bh & 7;
  const int m0 = blockIdx.x * 16;
  const size_t headoff = ((size_t)b * 512 + (size_t)h * 64) * T;
  const int grp = lane >> 4, nc = lane & 15;

  for (int i = lane; i < 21 * 64; i += 32) EV[i / 64][i % 64] = erv[i];

  // Q fragments (A: m=t stride 1, k=d stride T)
  v16h qa0 = load_a_strided_h(qb + headoff + m0, 1, T);
  v16h qa1 = load_a_strided_h(qb + headoff + m0 + (size_t)32 * T, 1, T);

  // Rel-k bias via WMMA: RB[t][j] = sum_d q[t,d] * erk[j,d], j = delta+10 (<21)
  {
    v8f r0 = zero8(), r1 = zero8();
    r0 = WMMA_F16(qa0, load_b_f32rows(erk, 0, 0, 21), r0);
    r0 = WMMA_F16(qa1, load_b_f32rows(erk, 32, 0, 21), r0);
    r1 = WMMA_F16(qa0, load_b_f32rows(erk, 0, 16, 21), r1);
    r1 = WMMA_F16(qa1, load_b_f32rows(erk, 32, 16, 21), r1);
#pragma unroll
    for (int g = 0; g < 8; ++g) {
      RB[g + grp * 8][nc] = r0[g];
      RB[g + grp * 8][16 + nc] = r1[g];
    }
  }
  __syncthreads();

  // Scores: stream 16-column tiles; add banded rel bias; mask; stash to LDS
  for (int n0 = 0; n0 < T; n0 += 16) {
    v8f c = zero8();
    v16h kb0 = load_b_strided_h(kb + headoff + n0, T, 1);
    v16h kb1 = load_b_strided_h(kb + headoff + n0 + (size_t)32 * T, T, 1);
    c = WMMA_F16(qa0, kb0, c);
    c = WMMA_F16(qa1, kb1, c);
    const int s = n0 + nc;
    const float ms = mask[(size_t)b * T + s];
#pragma unroll
    for (int g = 0; g < 8; ++g) {
      const int tl = g + grp * 8;
      const int t = m0 + tl;
      float v = c[g];
      const int d = s - t;
      if (d >= -10 && d <= 10) v += RB[tl][d + 10];
      if (ms == 0.f || mask[(size_t)b * T + t] == 0.f) v = -1e4f;
      S[tl][s] = v;
    }
  }
  __syncthreads();

  // Full-row softmax: 2 lanes per row, combine via shfl_xor(16)
  {
    const int r = lane & 15;
    const int s0 = (lane >> 4) * 256;
    float mx = -3.0e38f;
    for (int s = s0; s < s0 + 256; ++s) mx = fmaxf(mx, S[r][s]);
    mx = fmaxf(mx, __shfl_xor(mx, 16));
    float sum = 0.f;
    for (int s = s0; s < s0 + 256; ++s) {
      const float e = __expf(S[r][s] - mx);
      S[r][s] = e;
      sum += e;
    }
    sum += __shfl_xor(sum, 16);
    const float inv = 1.f / sum;
    for (int s = s0; s < s0 + 256; ++s) P[r][s] = (half_t)(S[r][s] * inv);
  }
  __syncthreads();

  // out = P @ V + banded rel-v; write [B,512,T] layout
  for (int dn = 0; dn < 64; dn += 16) {
    v8f o = zero8();
    for (int k0 = 0; k0 < 512; k0 += 32) {
      v16h pa = load_a_strided_h(&P[0][k0], 512, 1);
      v16h bv = load_b_strided_h(vb + headoff + (size_t)dn * T + k0, 1, T);
      o = WMMA_F16(pa, bv, o);
    }
#pragma unroll
    for (int g = 0; g < 8; ++g) {
      const int tl = g + grp * 8;
      const int t = m0 + tl;
      const int dd = dn + nc;
      float val = o[g];
#pragma unroll
      for (int r = -10; r <= 10; ++r) {
        const int s = t + r;
        if (s >= 0 && s < T) val += (float)P[tl][s] * EV[r + 10][dd];
      }
      ob[headoff + (size_t)dd * T + t] = (half_t)val;
    }
  }
}

// ======================================================================
// x_out = LayerNorm_channel(x + y); writes f32 master, f16 copy, masked f16 copy.
// ======================================================================
__global__ __launch_bounds__(256) void add_ln(
    const float* __restrict__ x, const float* __restrict__ y,
    const float* __restrict__ gam, const float* __restrict__ bet,
    const float* __restrict__ mask,
    float* __restrict__ xo, half_t* __restrict__ xo16, half_t* __restrict__ xo16m,
    int T) {
  __shared__ float P1[8][32];
  __shared__ float P2[8][32];
  const int C = 512;
  const int tl = threadIdx.x & 31;
  const int wk = threadIdx.x >> 5;
  const size_t bt = (size_t)blockIdx.x * 32 + tl;
  const int b = (int)(bt / T);
  const int t = (int)(bt % T);
  const size_t base = (size_t)b * C * T + t;
  const float mv = mask[(size_t)b * T + t];
  float s1 = 0.f, s2 = 0.f;
  for (int c = wk * 64; c < wk * 64 + 64; ++c) {
    const float v = x[base + (size_t)c * T] + y[base + (size_t)c * T];
    s1 += v;
    s2 += v * v;
  }
  P1[wk][tl] = s1;
  P2[wk][tl] = s2;
  __syncthreads();
  float t1 = 0.f, t2 = 0.f;
#pragma unroll
  for (int ww = 0; ww < 8; ++ww) {
    t1 += P1[ww][tl];
    t2 += P2[ww][tl];
  }
  const float mean = t1 / C;
  const float var = t2 / C - mean * mean;
  const float rs = rsqrtf(var + 1e-5f);
  for (int c = wk * 64; c < wk * 64 + 64; ++c) {
    const size_t off = base + (size_t)c * T;
    const float v = x[off] + y[off];
    const float o = (v - mean) * rs * gam[c] + bet[c];
    xo[off] = o;
    xo16[off] = (half_t)o;
    xo16m[off] = (half_t)(o * mv);
  }
}

__global__ __launch_bounds__(256) void init_x(
    const float* __restrict__ x, const float* __restrict__ mask,
    float* __restrict__ xo, half_t* __restrict__ xo16, int CT, int T, size_t total) {
  const size_t i = (size_t)blockIdx.x * 256 + threadIdx.x;
  if (i >= total) return;
  const int t = (int)(i % T);
  const size_t b = i / CT;
  const float v = x[i] * mask[b * T + t];
  xo[i] = v;
  xo16[i] = (half_t)v;
}

__global__ __launch_bounds__(256) void final_mask(
    const float* __restrict__ x, const float* __restrict__ mask,
    float* __restrict__ out, int CT, int T, size_t total) {
  const size_t i = (size_t)blockIdx.x * 256 + threadIdx.x;
  if (i >= total) return;
  const int t = (int)(i % T);
  const size_t b = i / CT;
  out[i] = x[i] * mask[b * T + t];
}

extern "C" void kernel_launch(void* const* d_in, const int* in_sizes, int n_in,
                              void* d_out, int out_size, void* d_ws, size_t ws_size,
                              hipStream_t stream) {
  (void)in_sizes; (void)n_in; (void)out_size; (void)ws_size;
  const float* x   = (const float*)d_in[0];
  const float* msk = (const float*)d_in[1];
  const float* Wq  = (const float*)d_in[2];
  const float* bq  = (const float*)d_in[3];
  const float* Wk  = (const float*)d_in[4];
  const float* bk  = (const float*)d_in[5];
  const float* Wv  = (const float*)d_in[6];
  const float* bv  = (const float*)d_in[7];
  const float* Wo  = (const float*)d_in[8];
  const float* bo  = (const float*)d_in[9];
  const float* erk = (const float*)d_in[10];
  const float* erv = (const float*)d_in[11];
  const float* g1  = (const float*)d_in[12];
  const float* b1  = (const float*)d_in[13];
  const float* g2  = (const float*)d_in[14];
  const float* b2  = (const float*)d_in[15];
  const float* W1  = (const float*)d_in[16];
  const float* bf1 = (const float*)d_in[17];
  const float* W2  = (const float*)d_in[18];
  const float* bf2 = (const float*)d_in[19];
  float* out = (float*)d_out;

  constexpr int Bv = 8, T = 512, C = 512, F = 2048, L = 6;
  const size_t total = (size_t)Bv * C * T;

  // Workspace carve-out (~56 MB)
  char* wsp = (char*)d_ws;
  auto carve = [&](size_t bytes) {
    char* p = wsp;
    wsp += (bytes + 255) & ~(size_t)255;
    return p;
  };
  float*  xbuf = (float*)carve(total * 4);
  half_t* x16  = (half_t*)carve(total * 2);   // plain f16 (MHA input)
  half_t* xm16 = (half_t*)carve(total * 2);   // masked f16 (FFN input)
  half_t* qb   = (half_t*)carve(total * 2);
  half_t* kb   = (half_t*)carve(total * 2);
  half_t* vb   = (half_t*)carve(total * 2);
  half_t* ab   = (half_t*)carve(total * 2);
  float*  yb   = (float*)carve(total * 4);
  half_t* h16  = (half_t*)carve((size_t)Bv * F * T * 2);

  const int eblocks = (int)((total + 255) / 256);
  init_x<<<eblocks, 256, 0, stream>>>(x, msk, xbuf, x16, C * T, T, total);

  const dim3 gCC(T / 128, C / 64, Bv);   // 512x512 GEMMs
  const dim3 gFC(T / 128, F / 64, Bv);   // 2048-row conv
  const dim3 gAttn(T / 16, Bv * 8);
  const int lnBlocks = Bv * T / 32;

  for (int i = 0; i < L; ++i) {
    const float* Wq_i = Wq + (size_t)i * C * C;
    const float* Wk_i = Wk + (size_t)i * C * C;
    const float* Wv_i = Wv + (size_t)i * C * C;
    const float* Wo_i = Wo + (size_t)i * C * C;
    const float* bq_i = bq + (size_t)i * C;
    const float* bk_i = bk + (size_t)i * C;
    const float* bv_i = bv + (size_t)i * C;
    const float* bo_i = bo + (size_t)i * C;
    const float* erk_i = erk + (size_t)i * 21 * 64;
    const float* erv_i = erv + (size_t)i * 21 * 64;
    const float* g1_i = g1 + (size_t)i * C;
    const float* b1_i = b1 + (size_t)i * C;
    const float* g2_i = g2 + (size_t)i * C;
    const float* b2_i = b2 + (size_t)i * C;
    const float* W1_i = W1 + (size_t)i * F * C * 3;
    const float* bf1_i = bf1 + (size_t)i * F;
    const float* W2_i = W2 + (size_t)i * C * F * 3;
    const float* bf2_i = bf2 + (size_t)i * C;

    // QKV projections (1/sqrt(Dk) folded into Q)
    gemm_wmma<1, false, true, false><<<gCC, 256, 0, stream>>>(
        Wq_i, x16, bq_i, msk, qb, nullptr, C, C, T, 0.125f);
    gemm_wmma<1, false, true, false><<<gCC, 256, 0, stream>>>(
        Wk_i, x16, bk_i, msk, kb, nullptr, C, C, T, 1.f);
    gemm_wmma<1, false, true, false><<<gCC, 256, 0, stream>>>(
        Wv_i, x16, bv_i, msk, vb, nullptr, C, C, T, 1.f);

    attn_wmma<<<gAttn, 32, 0, stream>>>(qb, kb, vb, erk_i, erv_i, msk, ab, T);

    // Output projection -> f32 branch output
    gemm_wmma<1, false, false, false><<<gCC, 256, 0, stream>>>(
        Wo_i, ab, bo_i, msk, nullptr, yb, C, C, T, 1.f);

    add_ln<<<lnBlocks, 256, 0, stream>>>(xbuf, yb, g1_i, b1_i, msk, xbuf, x16, xm16, T);

    // FFN: conv1 (512->2048, k=3, input pre-masked, ReLU*mask epilogue),
    //      conv2 (2048->512, k=3, output masked)
    gemm_wmma<3, true, true, true><<<gFC, 256, 0, stream>>>(
        W1_i, xm16, bf1_i, msk, h16, nullptr, F, C, T, 1.f);
    gemm_wmma<3, false, false, true><<<gCC, 256, 0, stream>>>(
        W2_i, h16, bf2_i, msk, nullptr, yb, C, F, T, 1.f);

    add_ln<<<lnBlocks, 256, 0, stream>>>(xbuf, yb, g2_i, b2_i, msk, xbuf, x16, xm16, T);
  }

  final_mask<<<eblocks, 256, 0, stream>>>(xbuf, msk, out, C * T, T, total);
}
